// RAN_30202210025883
// MI455X (gfx1250) — compile-verified, matrix operand order checked by reading, not run on hardware
//
#include <hip/hip_runtime.h>

// ---------------------------------------------------------------------------
// RAN (Recurrent Additive Network) for MI455X / gfx1250 (wave32, WMMA).
//
//   Phase 0: cast x and all 5 weight matrices f32 -> bf16 (RNE).
//   Phase 1: fused triple GEMM (ctilde/ix/fx), M=32768, N=1024, K=1024.
//            Tiles staged Global->LDS with global_load_async_to_lds_b128
//            (ASYNCcnt, double buffered), fragments read via ds_load_b128,
//            math on v_wmma_f32_16x16x32_bf16 with f32 accumulators.
//            Streaming outputs stored non-temporally (TH_NT) to keep the
//            L2-resident x / weights hot.
//   Phase 2: 512 sequential step kernels. Each wave owns one 16x16 tile and
//            BOTH gate accumulators -> register-resident fused epilogue:
//              i = sigmoid(c@Wic^T + bic + ixa_t)
//              f = sigmoid(c@Wfc^T + bfc + fxa_t)
//              c = i*ctilde + f*c ;  h = tanh(c)
//            Recurrent weights (4 MB) stay resident in the 192 MB L2.
// ---------------------------------------------------------------------------

typedef __attribute__((ext_vector_type(16))) __bf16        bf16x16;
typedef __attribute__((ext_vector_type(8)))  float         f32x8;
typedef __attribute__((ext_vector_type(4)))  float         f32x4;
typedef __attribute__((ext_vector_type(4)))  unsigned int  u32x4;
typedef __attribute__((ext_vector_type(2)))  unsigned int  u32x2;

constexpr int Td = 512, Bd = 64, Id = 1024, Hd = 1024;
constexpr int Md = Td * Bd;   // 32768 GEMM rows
constexpr int Kd = 1024;      // == Id == Hd
constexpr int BK = 32;        // K-slice per pipeline stage (one WMMA K)
constexpr int LDS_LD = 40;    // 32 + 8 pad (80B row: 16B aligned, 20-dword
                              // stride spreads 16 rows over 16 distinct banks)

__device__ __forceinline__ unsigned short f2bf(float f) {
  unsigned int u = __float_as_uint(f);
  u += 0x7FFFu + ((u >> 16) & 1u);          // round-to-nearest-even
  return (unsigned short)(u >> 16);
}

__device__ __forceinline__ f32x8 wmma_bf16(bf16x16 a, bf16x16 b, f32x8 c) {
  // (neg_a, A, neg_b, B, c_mod, C, reuse_a, reuse_b)
  return __builtin_amdgcn_wmma_f32_16x16x32_bf16(false, a, false, b,
                                                 (short)0, c, false, false);
}

__device__ __forceinline__ float sigmoid_f(float x) {
  return 1.0f / (1.0f + __expf(-x));
}

__device__ __forceinline__ float tanh_f(float x) {
#if __has_builtin(__builtin_amdgcn_tanhf)
  return __builtin_amdgcn_tanhf(x);         // v_tanh_f32 (gfx1250 TRANS op)
#else
  float e = __expf(2.0f * x);
  return (e - 1.0f) / (e + 1.0f);
#endif
}

// Async 16B copy Global -> LDS (per-lane), tracked by ASYNCcnt.
__device__ __forceinline__ void async_b128(const void* g, unsigned lds_off) {
  asm volatile("global_load_async_to_lds_b128 %0, %1, off"
               :: "v"(lds_off), "v"((unsigned long long)g)
               : "memory");
}

__device__ __forceinline__ unsigned lds_off32(const void* p) {
  // generic LDS pointer: addr[31:0] is the LDS byte offset (ISA 10.2 aperture)
  return (unsigned)(unsigned long long)p;
}

// 16-bit A/B fragment per ISA 7.12.2: lane holds row lane%16; two contiguous
// 16B chunks at (lane>>4)*8 and 16 + (lane>>4)*8 within the 32-wide K slice.
__device__ __forceinline__ bf16x16 frag_global(const unsigned short* __restrict__ p,
                                               int ld, int row, int k0, int grp) {
  const unsigned short* q = p + (size_t)row * ld + k0 + grp * 8;
  union { u32x4 u[2]; bf16x16 v; } t;
  t.u[0] = *(const u32x4*)(q);
  t.u[1] = *(const u32x4*)(q + 16);
  return t.v;
}

__device__ __forceinline__ bf16x16 frag_lds(const unsigned short* p, int row, int grp) {
  const unsigned short* q = p + row * LDS_LD + grp * 8;
  union { u32x4 u[2]; bf16x16 v; } t;
  t.u[0] = *(const u32x4*)(q);        // ds_load_b128
  t.u[1] = *(const u32x4*)(q + 16);   // ds_load_b128
  return t.v;
}

// ---------------------------------------------------------------------------
// Phase 0 kernels: f32 -> bf16 casts
// ---------------------------------------------------------------------------
__global__ __launch_bounds__(256) void cvt_f32_to_bf16(
    const float* __restrict__ in, unsigned short* __restrict__ out, int n4) {
  int i = blockIdx.x * 256 + threadIdx.x;
  if (i >= n4) return;
  f32x4 v = ((const f32x4*)in)[i];
  u32x2 o;
  o.x = (unsigned)f2bf(v.x) | ((unsigned)f2bf(v.y) << 16);
  o.y = (unsigned)f2bf(v.z) | ((unsigned)f2bf(v.w) << 16);
  ((u32x2*)out)[i] = o;
}

__global__ __launch_bounds__(256) void init_c(
    const float* __restrict__ h0, float* __restrict__ c32,
    unsigned short* __restrict__ cb, int n) {
  int i = blockIdx.x * 256 + threadIdx.x;
  if (i >= n) return;
  float v = h0[i];
  c32[i] = v;
  cb[i]  = f2bf(v);
}

// ---------------------------------------------------------------------------
// Phase 1: fused triple GEMM.  grid = (M/128, N/128, 3), block = 256 (8 waves)
// WG tile 128x128; wave (2x4) computes 64x32 C = 4x2 WMMA tiles.
// A/B K-slices staged to LDS by async copies, double buffered.
// ---------------------------------------------------------------------------
__device__ __forceinline__ void issue_stage(
    const unsigned short* __restrict__ A, const unsigned short* __restrict__ W,
    unsigned short* ldsA, unsigned short* ldsB,
    int blkRow, int blkCol, int k0, int tid) {
  // 256 threads move 128 rows x 64B for each of A and B: row = tid>>1,
  // each thread copies two 16B chunks per matrix -> 4 async instr / thread.
  const int r  = tid >> 1;
  const int c0 = (tid & 1) * 2;               // chunk index 0..3 (16B chunks)
  const unsigned short* ga = A + (size_t)(blkRow + r) * Kd + k0 + c0 * 8;
  const unsigned short* gb = W + (size_t)(blkCol + r) * Kd + k0 + c0 * 8;
  const unsigned la = lds_off32(ldsA + r * LDS_LD + c0 * 8);
  const unsigned lb = lds_off32(ldsB + r * LDS_LD + c0 * 8);
  async_b128(ga,     la);
  async_b128(ga + 8, la + 16);
  async_b128(gb,     lb);
  async_b128(gb + 8, lb + 16);
}

__global__ __launch_bounds__(256) void ran_gemm3(
    const unsigned short* __restrict__ A,     // [M,K] bf16
    const unsigned short* __restrict__ w0, const unsigned short* __restrict__ w1,
    const unsigned short* __restrict__ w2,    // [N,K] bf16 each
    const float* __restrict__ b0, const float* __restrict__ b1,
    const float* __restrict__ b2,
    float* __restrict__ y0, float* __restrict__ y1, float* __restrict__ y2) {
  __shared__ unsigned short tA[2][128 * LDS_LD];
  __shared__ unsigned short tB[2][128 * LDS_LD];

  const unsigned short* W; const float* bias; float* Y;
  if (blockIdx.z == 0)      { W = w0; bias = b0; Y = y0; }
  else if (blockIdx.z == 1) { W = w1; bias = b1; Y = y1; }
  else                      { W = w2; bias = b2; Y = y2; }

  const int tid  = threadIdx.x;
  const int lane = tid & 31, wid = tid >> 5;
  const int waveM = wid >> 2, waveN = wid & 3;
  const int lrow = lane & 15, grp = lane >> 4;
  const int blkRow = blockIdx.x * 128;
  const int blkCol = blockIdx.y * 128;

  f32x8 acc[4][2] = {};

  issue_stage(A, W, &tA[0][0], &tB[0][0], blkRow, blkCol, 0, tid);

  constexpr int nStages = Kd / BK;            // 32
#pragma unroll 2
  for (int s = 0; s < nStages; ++s) {
    const unsigned short* curA = &tA[s & 1][0];
    const unsigned short* curB = &tB[s & 1][0];
    if (s + 1 < nStages) {
      issue_stage(A, W, &tA[(s + 1) & 1][0], &tB[(s + 1) & 1][0],
                  blkRow, blkCol, (s + 1) * BK, tid);
      // 4 async ops just issued for the next stage may stay in flight;
      // wait only for this stage's 4 (in-order completion).
      asm volatile("s_wait_asynccnt 0x4" ::: "memory");
    } else {
      asm volatile("s_wait_asynccnt 0x0" ::: "memory");
    }
    __syncthreads();

    bf16x16 a[4], b[2];
#pragma unroll
    for (int i = 0; i < 4; ++i)
      a[i] = frag_lds(curA, waveM * 64 + i * 16 + lrow, grp);
#pragma unroll
    for (int j = 0; j < 2; ++j)
      b[j] = frag_lds(curB, waveN * 32 + j * 16 + lrow, grp);
#pragma unroll
    for (int i = 0; i < 4; ++i)
#pragma unroll
      for (int j = 0; j < 2; ++j)
        acc[i][j] = wmma_bf16(a[i], b[j], acc[i][j]);

    __syncthreads();   // all reads done before this buffer is refilled
  }

  // Epilogue: C/D layout -> VGPR r holds row (r + 8*grp), col = lane%16.
  // Outputs are streamed (read once much later) -> non-temporal stores.
#pragma unroll
  for (int j = 0; j < 2; ++j) {
    int col = blkCol + waveN * 32 + j * 16 + lrow;
    float bv = bias[col];
#pragma unroll
    for (int i = 0; i < 4; ++i) {
#pragma unroll
      for (int r = 0; r < 8; ++r) {
        int row = blkRow + waveM * 64 + i * 16 + r + 8 * grp;
        __builtin_nontemporal_store(acc[i][j][r] + bv, &Y[(size_t)row * Hd + col]);
      }
    }
  }
}

// ---------------------------------------------------------------------------
// Phase 2: one recurrent step.  grid = 32, block = 256 (8 waves).
// wave w: mTile = w&3 (batch rows), nTile = blockIdx.x*32 + (w>>2)*16.
// Direct global fragment loads: c (128 KB) + Wic/Wfc (4 MB) are L2-resident.
// ---------------------------------------------------------------------------
__global__ __launch_bounds__(256) void ran_step(
    const unsigned short* __restrict__ cb_in,   // [B,H] bf16  (c_{t-1})
    const float* __restrict__ c_in,             // [B,H] f32   (c_{t-1})
    unsigned short* __restrict__ cb_out,        // [B,H] bf16  (c_t)
    float* __restrict__ c_out,                  // [B,H] f32   (c_t)
    const unsigned short* __restrict__ Wic,     // [H,H] bf16
    const unsigned short* __restrict__ Wfc,     // [H,H] bf16
    const float* __restrict__ bic, const float* __restrict__ bfc,
    const float* __restrict__ ixa_t,            // [B,H] f32 (streamed)
    const float* __restrict__ fxa_t,            // [B,H] f32 (streamed)
    float* hct) {                               // [B,H]: reads ctilde, writes h
  const int lane = threadIdx.x & 31, wid = threadIdx.x >> 5;
  const int mTile = wid & 3, np = wid >> 2;
  const int lrow = lane & 15, grp = lane >> 4;
  const int rowBase = mTile * 16;
  const int colTile = blockIdx.x * 32 + np * 16;
  const int col = colTile + lrow;

  f32x8 ai = {}, af = {};
  for (int k0 = 0; k0 < Hd; k0 += 32) {
    bf16x16 a  = frag_global(cb_in, Hd, rowBase + lrow, k0, grp);
    bf16x16 bi = frag_global(Wic,   Hd, colTile + lrow, k0, grp);
    bf16x16 bf = frag_global(Wfc,   Hd, colTile + lrow, k0, grp);
    ai = wmma_bf16(a, bi, ai);
    af = wmma_bf16(a, bf, af);
  }

  const float bi_v = bic[col], bf_v = bfc[col];
#pragma unroll
  for (int r = 0; r < 8; ++r) {
    int row = rowBase + r + 8 * grp;            // batch index 0..63
    int idx = row * Hd + col;
    float iv = sigmoid_f(ai[r] + bi_v + __builtin_nontemporal_load(&ixa_t[idx]));
    float fv = sigmoid_f(af[r] + bf_v + __builtin_nontemporal_load(&fxa_t[idx]));
    float ct = __builtin_nontemporal_load(&hct[idx]);
    float cn = iv * ct + fv * c_in[idx];
    c_out[idx]  = cn;                           // re-read next step: keep RT
    cb_out[idx] = f2bf(cn);
    __builtin_nontemporal_store(tanh_f(cn), &hct[idx]);  // h never re-read
  }
}

// ---------------------------------------------------------------------------
// Host launcher
// ---------------------------------------------------------------------------
extern "C" void kernel_launch(void* const* d_in, const int* in_sizes, int n_in,
                              void* d_out, int out_size, void* d_ws, size_t ws_size,
                              hipStream_t stream) {
  const float* x   = (const float*)d_in[0];
  const float* h0  = (const float*)d_in[1];
  const float* wcx = (const float*)d_in[2];
  const float* wic = (const float*)d_in[3];
  const float* wix = (const float*)d_in[4];
  const float* wfc = (const float*)d_in[5];
  const float* wfx = (const float*)d_in[6];
  const float* bcx = (const float*)d_in[7];
  const float* bic = (const float*)d_in[8];
  const float* bix = (const float*)d_in[9];
  const float* bfc = (const float*)d_in[10];
  const float* bfx = (const float*)d_in[11];
  float* out = (float*)d_out;          // hs [T,B,H] then c_T [B,H]

  // ---- workspace carve-up (256B aligned) ----
  char* ws = (char*)d_ws;
  size_t off = 0;
  auto carve = [&](size_t bytes) -> void* {
    void* p = ws + off;
    off += (bytes + 255) & ~(size_t)255;
    return p;
  };
  unsigned short* xb    = (unsigned short*)carve((size_t)Md * Id * 2);
  unsigned short* wcx_b = (unsigned short*)carve((size_t)Hd * Id * 2);
  unsigned short* wix_b = (unsigned short*)carve((size_t)Hd * Id * 2);
  unsigned short* wfx_b = (unsigned short*)carve((size_t)Hd * Id * 2);
  unsigned short* wic_b = (unsigned short*)carve((size_t)Hd * Hd * 2);
  unsigned short* wfc_b = (unsigned short*)carve((size_t)Hd * Hd * 2);
  float* ixa = (float*)carve((size_t)Md * Hd * 4);
  float* fxa = (float*)carve((size_t)Md * Hd * 4);
  float* c32[2];
  unsigned short* cb16[2];
  c32[0]  = (float*)carve((size_t)Bd * Hd * 4);
  c32[1]  = (float*)carve((size_t)Bd * Hd * 4);
  cb16[0] = (unsigned short*)carve((size_t)Bd * Hd * 2);
  cb16[1] = (unsigned short*)carve((size_t)Bd * Hd * 2);

  // ---- phase 0: casts ----
  {
    int n4 = Md * Id / 4;
    cvt_f32_to_bf16<<<(n4 + 255) / 256, 256, 0, stream>>>(x, xb, n4);
    int w4 = Hd * Id / 4;
    cvt_f32_to_bf16<<<(w4 + 255) / 256, 256, 0, stream>>>(wcx, wcx_b, w4);
    cvt_f32_to_bf16<<<(w4 + 255) / 256, 256, 0, stream>>>(wix, wix_b, w4);
    cvt_f32_to_bf16<<<(w4 + 255) / 256, 256, 0, stream>>>(wfx, wfx_b, w4);
    cvt_f32_to_bf16<<<(w4 + 255) / 256, 256, 0, stream>>>(wic, wic_b, w4);
    cvt_f32_to_bf16<<<(w4 + 255) / 256, 256, 0, stream>>>(wfc, wfc_b, w4);
    int nc = Bd * Hd;
    init_c<<<(nc + 255) / 256, 256, 0, stream>>>(h0, c32[0], cb16[0], nc);
  }

  // ---- phase 1: ctilde -> d_out hs region, ixa/fxa -> workspace ----
  ran_gemm3<<<dim3(Md / 128, Hd / 128, 3), 256, 0, stream>>>(
      xb, wcx_b, wix_b, wfx_b, bcx, bix, bfx, out, ixa, fxa);

  // ---- phase 2: 512 recurrent steps ----
  for (int t = 0; t < Td; ++t) {
    int p = t & 1;
    float* hct = out + (size_t)t * Bd * Hd;   // ctilde in, h out (in place)
    ran_step<<<Hd / 32, 256, 0, stream>>>(
        cb16[p], c32[p], cb16[p ^ 1], c32[p ^ 1],
        wic_b, wfc_b, bic, bfc,
        ixa + (size_t)t * Bd * Hd, fxa + (size_t)t * Bd * Hd, hct);
  }

  // final c lives in buffer 0 after 512 steps (even count)
  hipMemcpyAsync(out + (size_t)Md * Hd, c32[0], (size_t)Bd * Hd * sizeof(float),
                 hipMemcpyDeviceToDevice, stream);
}
